// Net_45629732553073
// MI455X (gfx1250) — compile-verified, matrix operand order checked by reading, not run on hardware
//
#include <hip/hip_runtime.h>
#include <hip/hip_bf16.h>

// ---------------------------------------------------------------------------
// Sizes (fixed by the reference)
// ---------------------------------------------------------------------------
#define B_N   16384       // batch
#define D_K   3072        // C*H*W reduction dim
#define K_P   300         // prototypes
#define KPAD  320         // 300 padded to 20 WMMA N-tiles
#define NCLS  10

typedef __attribute__((ext_vector_type(16))) __bf16 v16bf;
typedef __attribute__((ext_vector_type(8)))  float  v8f;

// fp32 -> bf16 round-to-nearest-even (integer ops only)
__device__ __forceinline__ unsigned short f2bf(float x) {
    unsigned u = __float_as_uint(x);
    u += 0x7FFFu + ((u >> 16) & 1u);
    return (unsigned short)(u >> 16);
}
__device__ __forceinline__ float bf2f(unsigned short h) {
    return __uint_as_float(((unsigned)h) << 16);
}

// Split 8 fp32 -> 8 bf16 hi (RNE) + 8 bf16 lo (trunc), packed as uint4 each.
__device__ __forceinline__ void split8(const float4 a, const float4 b,
                                       uint4& hq, uint4& lq) {
    float v[8] = {a.x, a.y, a.z, a.w, b.x, b.y, b.z, b.w};
    union { uint4 q; unsigned short u[8]; } H, L;
    #pragma unroll
    for (int j = 0; j < 8; ++j) {
        const unsigned short h = f2bf(v[j]);
        H.u[j] = h;
        const float lo = v[j] - bf2f(h);
        L.u[j] = (unsigned short)(__float_as_uint(lo) >> 16);   // trunc
    }
    hq = H.q; lq = L.q;
}

// ---------------------------------------------------------------------------
// 1) Row norms of x: one wave per row, coalesced float4 loads
// ---------------------------------------------------------------------------
__global__ __launch_bounds__(256) void rownorm_x(const float* __restrict__ x,
                                                 float* __restrict__ rnx) {
    const int wave = threadIdx.x >> 5;
    const int lane = threadIdx.x & 31;
    const int row  = blockIdx.x * 8 + wave;
    const float4* p = (const float4*)(x + (size_t)row * D_K);
    float s = 0.f;
    #pragma unroll 4
    for (int i = 0; i < D_K / 128; ++i) {
        float4 v = p[i * 32 + lane];
        s += v.x * v.x + v.y * v.y + v.z * v.z + v.w * v.w;
    }
    for (int m = 16; m; m >>= 1) s += __shfl_xor(s, m, 32);
    if (lane == 0) rnx[row] = 1.f / (sqrtf(s) + 1e-12f);
}

// 2) Row norms of w5 (300 rows)
__global__ __launch_bounds__(256) void rownorm_w(const float* __restrict__ w5,
                                                 float* __restrict__ rnw) {
    const int wave = threadIdx.x >> 5;
    const int lane = threadIdx.x & 31;
    const int row  = blockIdx.x * 8 + wave;
    if (row >= K_P) return;
    const float4* p = (const float4*)(w5 + (size_t)row * D_K);
    float s = 0.f;
    for (int i = 0; i < D_K / 128; ++i) {
        float4 v = p[i * 32 + lane];
        s += v.x * v.x + v.y * v.y + v.z * v.z + v.w * v.w;
    }
    for (int m = 16; m; m >>= 1) s += __shfl_xor(s, m, 32);
    if (lane == 0) rnw[row] = 1.f / (sqrtf(s) + 1e-12f);
}

// 3) Normalize w5, split into bf16 hi/lo, pad rows 300..319 with zeros
__global__ __launch_bounds__(256) void wconv(const float* __restrict__ w5,
                                             const float* __restrict__ rnw,
                                             unsigned short* __restrict__ wh,
                                             unsigned short* __restrict__ wl) {
    const int idx = blockIdx.x * 256 + threadIdx.x;     // grid covers KPAD*D_K
    const int row = idx / D_K;
    const int col = idx - row * D_K;
    float v = 0.f;
    if (row < K_P) v = w5[(size_t)row * D_K + col] * rnw[row];
    const unsigned short h = f2bf(v);
    wh[idx] = h;
    wl[idx] = f2bf(v - bf2f(h));
}

// 4) Zero batch-norm accumulators (ws is poisoned once; must re-zero per call)
__global__ void zero_stats(float* __restrict__ cs, float* __restrict__ cq) {
    const int t = threadIdx.x;
    if (t < KPAD) { cs[t] = 0.f; cq[t] = 0.f; }
}

// ---------------------------------------------------------------------------
// 5) Main GEMM: fc5 = relu( (x . wn^T) * rnx ), bf16x2-split WMMA
//    Block: 256 thr (8 waves), BM=64 rows, all 320 cols, KC=64.
//    ~110 KB LDS via *dynamic* shared memory (gfx1250 allows 320 KB/WG;
//    the static-LDS compile-time cap doesn't apply). 60 WMMAs per barrier
//    pair per wave; B fragments double-buffered in registers so ds_load
//    latency of tile nt+1 hides under the WMMAs of tile nt.
// ---------------------------------------------------------------------------
#define BM  64
#define KC  64
#define LWS 72     // LDS row stride (ushorts) -> 144B: odd bank shift per row

#define LX_ELEMS   (BM * LWS)            // 4608 ushorts
#define LW_ELEMS   (KPAD * LWS)          // 23040 ushorts
#define SMEM_BYTES ((2 * LX_ELEMS + 2 * LW_ELEMS) * 2)   // 110,592 B

struct Frag { union { v16bf bf; uint4 q[2]; }; };

__global__ __launch_bounds__(256) void gemm_cos(const float* __restrict__ x,
                                                const unsigned short* __restrict__ wh,
                                                const unsigned short* __restrict__ wl,
                                                const float* __restrict__ rnx,
                                                float* __restrict__ fc5) {
    extern __shared__ unsigned short smem[];
    unsigned short* lxh = smem;                    // bf16-hi x tile
    unsigned short* lxl = lxh + LX_ELEMS;          // bf16-lo x tile
    unsigned short* lwh = lxl + LX_ELEMS;          // bf16-hi w tile
    unsigned short* lwl = lwh + LW_ELEMS;          // bf16-lo w tile

    const int tid  = threadIdx.x;
    const int lane = tid & 31;
    const int wave = tid >> 5;
    const int wm   = wave & 3;        // M tile 0..3
    const int wn   = wave >> 2;       // N half 0..1
    const int rowBase = blockIdx.x * BM;

    v8f acc[10];
    const v8f vzero = {0.f,0.f,0.f,0.f,0.f,0.f,0.f,0.f};
    #pragma unroll
    for (int i = 0; i < 10; ++i) acc[i] = vzero;

    const int ar = lane & 15;             // fragment row/col within tile
    const int ak = (lane >> 4) * 16;      // K half selected by lane group

    // staging coordinates: x has 512 8-float chunks (64 rows x 8), 2/thread
    const int xr  = tid >> 3;             // 0..31  (c>>3 pattern below)
    const int xc8 = (tid & 7) * 8;        // 0..56

    for (int k0 = 0; k0 < D_K; k0 += KC) {
        __syncthreads();
        // ---- stage x tile (64x64): load fp32, split to bf16 hi/lo in LDS
        #pragma unroll
        for (int i = 0; i < 2; ++i) {
            const int r = xr + i * 32;
            const float* gp = x + (size_t)(rowBase + r) * D_K + k0 + xc8;
            const float4 a = *(const float4*)(gp);
            const float4 b = *(const float4*)(gp + 4);
            uint4 hq, lq;
            split8(a, b, hq, lq);
            *(uint4*)(&lxh[r * LWS + xc8]) = hq;
            *(uint4*)(&lxl[r * LWS + xc8]) = lq;
        }
        // ---- stage w tiles (320x64 bf16 hi+lo): 2560 b128 chunks, 10/thread
        #pragma unroll
        for (int i = 0; i < 10; ++i) {
            const int c = tid + i * 256;
            const int r = c >> 3, c8 = (c & 7) * 8;
            const size_t g = (size_t)r * D_K + k0 + c8;
            *(uint4*)(&lwh[r * LWS + c8]) = *(const uint4*)(wh + g);
            *(uint4*)(&lwl[r * LWS + c8]) = *(const uint4*)(wl + g);
        }
        // prefetch next x chunk into L2 while we compute
        if (k0 + KC < D_K)
            __builtin_prefetch(x + (size_t)(rowBase + xr) * D_K + k0 + KC + xc8, 0, 0);
        __syncthreads();

        // ---- two 32-K sub-chunks per staged tile
        #pragma unroll
        for (int kk = 0; kk < KC; kk += 32) {
            const int ab = kk + ak;
            Frag Ah, Al;
            {
                const unsigned short* ph = &lxh[(wm * 16 + ar) * LWS + ab];
                const unsigned short* pl = &lxl[(wm * 16 + ar) * LWS + ab];
                Ah.q[0] = *(const uint4*)(ph);
                Ah.q[1] = *(const uint4*)(ph + 8);
                Al.q[0] = *(const uint4*)(pl);
                Al.q[1] = *(const uint4*)(pl + 8);
            }
            // B fragments: double-buffered so tile nt+1 loads issue before
            // the WMMAs of tile nt retire (no full dscnt drain per tile).
            Frag Bh[2], Bl[2];
            {
                const int col = wn * 160 + ar;
                const unsigned short* ph = &lwh[col * LWS + ab];
                const unsigned short* pl = &lwl[col * LWS + ab];
                Bh[0].q[0] = *(const uint4*)(ph);
                Bh[0].q[1] = *(const uint4*)(ph + 8);
                Bl[0].q[0] = *(const uint4*)(pl);
                Bl[0].q[1] = *(const uint4*)(pl + 8);
            }
            #pragma unroll
            for (int nt = 0; nt < 10; ++nt) {
                const int cur = nt & 1, nxt = cur ^ 1;
                if (nt < 9) {
                    const int col = wn * 160 + (nt + 1) * 16 + ar;
                    const unsigned short* ph = &lwh[col * LWS + ab];
                    const unsigned short* pl = &lwl[col * LWS + ab];
                    Bh[nxt].q[0] = *(const uint4*)(ph);
                    Bh[nxt].q[1] = *(const uint4*)(ph + 8);
                    Bl[nxt].q[0] = *(const uint4*)(pl);
                    Bl[nxt].q[1] = *(const uint4*)(pl + 8);
                }
                acc[nt] = __builtin_amdgcn_wmma_f32_16x16x32_bf16(
                    false, Ah.bf, false, Bh[cur].bf, (short)0, acc[nt], false, false);
                acc[nt] = __builtin_amdgcn_wmma_f32_16x16x32_bf16(
                    false, Al.bf, false, Bh[cur].bf, (short)0, acc[nt], false, false);
                acc[nt] = __builtin_amdgcn_wmma_f32_16x16x32_bf16(
                    false, Ah.bf, false, Bl[cur].bf, (short)0, acc[nt], false, false);
            }
        }
    }

    // ---- epilogue: scale by 1/(|x|+eps), ReLU, store fc5
    const int rOff  = (lane >> 4) * 8;    // C/D layout: lanes>=16 -> M += 8
    const int cLane = lane & 15;
    float rs[8];
    #pragma unroll
    for (int v = 0; v < 8; ++v)
        rs[v] = rnx[rowBase + wm * 16 + rOff + v];
    #pragma unroll
    for (int nt = 0; nt < 10; ++nt) {
        const int col = wn * 160 + nt * 16 + cLane;
        if (col < K_P) {
            #pragma unroll
            for (int v = 0; v < 8; ++v) {
                const int row = rowBase + wm * 16 + rOff + v;
                fc5[(size_t)row * K_P + col] = fmaxf(acc[nt][v] * rs[v], 0.f);
            }
        }
    }
}

// ---------------------------------------------------------------------------
// 6) Column sums / sums-of-squares over B (coalesced: tid == column)
// ---------------------------------------------------------------------------
__global__ void stats1(const float* __restrict__ fc5,
                       float* __restrict__ cs, float* __restrict__ cq) {
    const int t = threadIdx.x;
    if (t >= K_P) return;
    const int r0 = blockIdx.x * 64;
    float s = 0.f, q = 0.f;
    for (int r = 0; r < 64; ++r) {
        const float v = fc5[(size_t)(r0 + r) * K_P + t];
        s += v; q += v * v;
    }
    atomicAdd(&cs[t], s);
    atomicAdd(&cq[t], q);
}

// 7) Fold BN into per-channel affine: bn = a*v + b
__global__ void stats2(const float* __restrict__ cs, const float* __restrict__ cq,
                       const float* __restrict__ gamma, const float* __restrict__ beta,
                       float* __restrict__ ak, float* __restrict__ bk) {
    const int k = threadIdx.x;
    if (k >= K_P) return;
    const float mean = cs[k] * (1.f / (float)B_N);
    const float var  = cq[k] * (1.f / (float)B_N) - mean * mean;
    const float istd = rsqrtf(var + 1e-5f);
    const float a = gamma[k] * istd;
    ak[k] = a;
    bk[k] = beta[k] - mean * a;
}

// 8) bn5 + fc6 fused: one pass over fc5 per row, w6 cached in LDS
__global__ __launch_bounds__(256) void bnfc6(const float* __restrict__ fc5,
                                             const float* __restrict__ ak,
                                             const float* __restrict__ bk,
                                             const float* __restrict__ w6,
                                             float* __restrict__ bn5,
                                             float* __restrict__ fc6) {
    __shared__ float sa[K_P], sb[K_P], sw[NCLS * K_P];
    for (int i = threadIdx.x; i < K_P; i += 256) { sa[i] = ak[i]; sb[i] = bk[i]; }
    for (int i = threadIdx.x; i < NCLS * K_P; i += 256) sw[i] = w6[i];
    __syncthreads();

    const int row = blockIdx.x * 256 + threadIdx.x;
    const float* src = fc5 + (size_t)row * K_P;
    float*       dst = bn5 + (size_t)row * K_P;
    float f[NCLS];
    #pragma unroll
    for (int o = 0; o < NCLS; ++o) f[o] = 0.f;
    for (int k = 0; k < K_P; ++k) {
        const float bnv = sa[k] * src[k] + sb[k];
        dst[k] = bnv;
        #pragma unroll
        for (int o = 0; o < NCLS; ++o) f[o] += sw[o * K_P + k] * bnv;
    }
    #pragma unroll
    for (int o = 0; o < NCLS; ++o) fc6[(size_t)row * NCLS + o] = f[o];
}

// ---------------------------------------------------------------------------
extern "C" void kernel_launch(void* const* d_in, const int* in_sizes, int n_in,
                              void* d_out, int out_size, void* d_ws, size_t ws_size,
                              hipStream_t stream) {
    const float* x     = (const float*)d_in[0];
    const float* w5    = (const float*)d_in[1];
    const float* gamma = (const float*)d_in[2];
    const float* beta  = (const float*)d_in[3];
    const float* w6    = (const float*)d_in[4];

    float* out = (float*)d_out;
    float* fc5 = out;                                  // (B, 300)
    float* bn5 = out + (size_t)B_N * K_P;              // (B, 300)
    float* fc6 = out + 2ull * (size_t)B_N * K_P;       // (B, 10)

    // workspace layout (~4 MB total)
    unsigned short* wh  = (unsigned short*)d_ws;                 // KPAD*D_K bf16
    unsigned short* wl  = wh + (size_t)KPAD * D_K;               // KPAD*D_K bf16
    float* rnx    = (float*)(wl + (size_t)KPAD * D_K);           // B_N
    float* rnw    = rnx + B_N;                                   // KPAD
    float* colsum = rnw + KPAD;                                  // KPAD
    float* colsq  = colsum + KPAD;                               // KPAD
    float* ak     = colsq + KPAD;                                // KPAD
    float* bk     = ak + KPAD;                                   // KPAD

    hipLaunchKernelGGL(rownorm_x, dim3(B_N / 8), dim3(256), 0, stream, x, rnx);
    hipLaunchKernelGGL(rownorm_w, dim3((K_P + 7) / 8), dim3(256), 0, stream, w5, rnw);
    hipLaunchKernelGGL(wconv, dim3((KPAD * D_K) / 256), dim3(256), 0, stream,
                       w5, rnw, wh, wl);
    hipLaunchKernelGGL(zero_stats, dim3(1), dim3(KPAD), 0, stream, colsum, colsq);
    hipLaunchKernelGGL(gemm_cos, dim3(B_N / BM), dim3(256), SMEM_BYTES, stream,
                       x, wh, wl, rnx, fc5);
    hipLaunchKernelGGL(stats1, dim3(B_N / 64), dim3(320), 0, stream,
                       fc5, colsum, colsq);
    hipLaunchKernelGGL(stats2, dim3(1), dim3(KPAD), 0, stream,
                       colsum, colsq, gamma, beta, ak, bk);
    hipLaunchKernelGGL(bnfc6, dim3(B_N / 256), dim3(256), 0, stream,
                       fc5, ak, bk, w6, bn5, fc6);
}
